// MQALayer_60060822667336
// MI455X (gfx1250) — matrix operand order repaired and checked by reading, not compiled, not run on hardware
//
#include <hip/hip_runtime.h>
#include <hip/hip_bf16.h>

// ---------------------------------------------------------------------------
// MQA layer (DeepSeek-style) for gfx1250 / MI455X.
// All GEMM-like math goes through v_wmma_f32_16x16x32_bf16 (bf16 A/B, f32 acc).
// ---------------------------------------------------------------------------

typedef __attribute__((ext_vector_type(16))) __bf16 v16bf;
typedef __attribute__((ext_vector_type(8)))  float  v8f;

#define T_TOK   2048
#define HID     4096
#define NH      32
#define HD      192
#define DR      64
#define Q_LORA  1536
#define O_LORA  512
#define NG      4
#define EPSV    1e-6f
#define SCALEV  0.07216878364870322f   /* 192^-0.5 */

static __device__ __forceinline__ v8f wmma_bf16(v16bf a, v16bf b, v8f c) {
    // D = A(16x32 bf16) * B(32x16 bf16) + C(16x16 f32)
    return __builtin_amdgcn_wmma_f32_16x16x32_bf16(
        /*neg_a=*/false, a, /*neg_b=*/false, b,
        /*c_mod=*/(short)0, c, /*reuse_a=*/false, /*reuse_b=*/false);
}

// ---------------------------------------------------------------------------
// Generic GEMM: C[M,N] = A[M,K] * W[N,K]^T   (fp32 in/out, bf16 WMMA compute)
// One wave computes one 16x16 tile; block = 4 waves along N.
// Requires M%16==0, N%64==0, K%32==0 (true for every call site here).
// ---------------------------------------------------------------------------
__global__ __launch_bounds__(128)
void gemm_wmma_bf16(const float* __restrict__ A, const float* __restrict__ B,
                    float* __restrict__ C, int M, int N, int K,
                    int lda, int ldb, int ldc) {
    const int lane = threadIdx.x;          // 0..31
    const int half = lane >> 4;            // 0/1
    const int l16  = lane & 15;
    const int tileN = blockIdx.x * 4 + threadIdx.y;
    const int tileM = blockIdx.y;
    if (tileM * 16 >= M || tileN * 16 >= N) return;   // uniform per wave

    const float* Arow = A + (size_t)(tileM * 16 + l16) * lda;   // A row m = lane%16
    const float* Brow = B + (size_t)(tileN * 16 + l16) * ldb;   // B col n = lane%16

    v8f acc;
#pragma unroll
    for (int r = 0; r < 8; ++r) acc[r] = 0.0f;

    for (int kb = 0; kb < K; kb += 32) {
        if (kb + 32 < K) {
            __builtin_prefetch(Arow + kb + 32, 0, 0);
            __builtin_prefetch(Brow + kb + 32, 0, 0);
        }
        v16bf a, b;
        // A-fragment: element e=g*8+j  <->  K = kb + half*16 + g*8 + j
#pragma unroll
        for (int g = 0; g < 2; ++g)
#pragma unroll
            for (int j = 0; j < 8; ++j)
                a[g * 8 + j] = (__bf16)Arow[kb + half * 16 + g * 8 + j];
        // B-fragment: element e  <->  K = kb + half*16 + e   (value = W[n][k])
#pragma unroll
        for (int e = 0; e < 16; ++e)
            b[e] = (__bf16)Brow[kb + half * 16 + e];
        acc = wmma_bf16(a, b, acc);
    }

    // D layout: element r -> row = half*8 + r, col = lane%16
#pragma unroll
    for (int r = 0; r < 8; ++r)
        C[(size_t)(tileM * 16 + half * 8 + r) * ldc + tileN * 16 + l16] = acc[r];
}

// ---------------------------------------------------------------------------
// Row-wise RMSNorm over C columns (fp32, in place). w may be null (unweighted).
// ---------------------------------------------------------------------------
__global__ __launch_bounds__(256)
void rmsnorm_kernel(float* __restrict__ X, const float* __restrict__ w, int C) {
    __shared__ float red[256];
    const int row = blockIdx.x;
    float* xr = X + (size_t)row * C;
    float ss = 0.0f;
    for (int i = threadIdx.x; i < C; i += 256) { float v = xr[i]; ss += v * v; }
    red[threadIdx.x] = ss;
    __syncthreads();
    for (int s = 128; s > 0; s >>= 1) {
        if (threadIdx.x < s) red[threadIdx.x] += red[threadIdx.x + s];
        __syncthreads();
    }
    const float rinv = rsqrtf(red[0] / (float)C + EPSV);
    for (int i = threadIdx.x; i < C; i += 256) {
        float v = xr[i] * rinv;
        xr[i] = w ? v * w[i] : v;
    }
}

// ---------------------------------------------------------------------------
// Interleaved RoPE on the last DR (=64) elements of each row.
// positions == arange(T), so pos = row / rows_per_pos. 32 pairs -> 32 threads.
// ---------------------------------------------------------------------------
__global__ __launch_bounds__(32)
void rope_kernel(float* __restrict__ X, const float* __restrict__ ct,
                 const float* __restrict__ st, int rows_per_pos, int rowlen) {
    const int row = blockIdx.x;
    const int p   = threadIdx.x;                 // pair index 0..31
    const int pos = row / rows_per_pos;
    const float c = ct[pos * (DR / 2) + p];
    const float s = st[pos * (DR / 2) + p];
    float* xr = X + (size_t)row * rowlen + (rowlen - DR);
    const float a = xr[2 * p], b = xr[2 * p + 1];
    xr[2 * p]     = a * c - b * s;
    xr[2 * p + 1] = a * s + b * c;
}

// ---------------------------------------------------------------------------
// Flash-style causal MQA attention with sink, fully WMMA:
//   one wave per (16-token q tile, head); 32 kv columns per step:
//   12 score WMMAs (K = D = 192) + 12 PV WMMAs (12 d-tiles).
// Online softmax seeded with m = sink, l = 1 reproduces the reference's
// "softmax over [scores, sink] then drop sink column" exactly.
// qbuf: [T,H,D] fp32 (post-norm, post-RoPE); kvbuf: [T,D]; obuf: [T,H,D].
// ---------------------------------------------------------------------------
__global__ __launch_bounds__(128)
void attn_wmma_kernel(const float* __restrict__ qbuf,
                      const float* __restrict__ kvbuf,
                      const float* __restrict__ sink,
                      float* __restrict__ obuf) {
    const int lane = threadIdx.x;
    const int wave = threadIdx.y;
    const int half = lane >> 4;
    const int l16  = lane & 15;
    const int q0   = blockIdx.x * 16;
    const int head = blockIdx.y * 4 + wave;

    __shared__ float Pl[4][16][32];   // per-wave probs patch for layout re-stripe

    // Load q fragments once: 6 K-steps of 32 over D=192.
    v16bf qf[6];
#pragma unroll
    for (int t = 0; t < 6; ++t) {
        const float* qrow = qbuf + ((size_t)(q0 + l16) * NH + head) * HD + t * 32;
#pragma unroll
        for (int g = 0; g < 2; ++g)
#pragma unroll
            for (int j = 0; j < 8; ++j)
                qf[t][g * 8 + j] = (__bf16)qrow[half * 16 + g * 8 + j];
    }

    const float sv = sink[head];
    float m[8], l[8];
#pragma unroll
    for (int r = 0; r < 8; ++r) { m[r] = sv; l[r] = 1.0f; }

    v8f acc[12];
#pragma unroll
    for (int t = 0; t < 12; ++t)
#pragma unroll
        for (int r = 0; r < 8; ++r) acc[t][r] = 0.0f;

    const int sEnd = q0 + 16;                    // causal: cols <= q0+15
    for (int s0 = 0; s0 < sEnd; s0 += 32) {
        // ---- scores for 32 kv columns (two 16-col tiles) ----
        float stile[2][8];
#pragma unroll
        for (int c = 0; c < 2; ++c) {
            v8f sc;
#pragma unroll
            for (int r = 0; r < 8; ++r) sc[r] = 0.0f;
#pragma unroll
            for (int t = 0; t < 6; ++t) {
                v16bf b;
                const float* krow =
                    kvbuf + (size_t)(s0 + c * 16 + l16) * HD + t * 32 + half * 16;
#pragma unroll
                for (int e = 0; e < 16; ++e) b[e] = (__bf16)krow[e];
                sc = wmma_bf16(qf[t], b, sc);
            }
            const int scol = s0 + c * 16 + l16;
#pragma unroll
            for (int r = 0; r < 8; ++r) {
                const int srow = q0 + half * 8 + r;
                stile[c][r] = (scol <= srow) ? sc[r] * SCALEV : -__builtin_inff();
            }
        }

        // ---- online softmax update (rows live on 16-lane half-groups) ----
        float p0a[8], p1a[8];
#pragma unroll
        for (int r = 0; r < 8; ++r) {
            float mx = fmaxf(stile[0][r], stile[1][r]);
#pragma unroll
            for (int off = 1; off < 16; off <<= 1)
                mx = fmaxf(mx, __shfl_xor(mx, off));
            const float newm  = fmaxf(m[r], mx);
            const float resc  = __expf(m[r] - newm);
            const float p0    = __expf(stile[0][r] - newm);
            const float p1    = __expf(stile[1][r] - newm);
            float sum = p0 + p1;
#pragma unroll
            for (int off = 1; off < 16; off <<= 1)
                sum += __shfl_xor(sum, off);
            l[r] = l[r] * resc + sum;
            m[r] = newm;
#pragma unroll
            for (int t = 0; t < 12; ++t) acc[t][r] *= resc;
            p0a[r] = p0; p1a[r] = p1;
        }

        // ---- re-stripe probs C-layout -> A-layout through LDS ----
        __syncthreads();   // protect previous iteration's reads (uniform trips)
#pragma unroll
        for (int r = 0; r < 8; ++r) {
            Pl[wave][half * 8 + r][l16]      = p0a[r];
            Pl[wave][half * 8 + r][16 + l16] = p1a[r];
        }
        __syncthreads();
        v16bf af;
#pragma unroll
        for (int g = 0; g < 2; ++g)
#pragma unroll
            for (int j = 0; j < 8; ++j)
                af[g * 8 + j] = (__bf16)Pl[wave][l16][half * 16 + g * 8 + j];

        // ---- PV: acc[16x192] += P[16x32] * kv[32x192] ----
#pragma unroll
        for (int t = 0; t < 12; ++t) {
            v16bf b;
            const float* kcol =
                kvbuf + (size_t)(s0 + half * 16) * HD + t * 16 + l16;
#pragma unroll
            for (int e = 0; e < 16; ++e) b[e] = (__bf16)kcol[(size_t)e * HD];
            acc[t] = wmma_bf16(af, b, acc[t]);
        }
    }

    // ---- normalize and store: row = q0 + half*8 + r, d = t*16 + lane%16 ----
#pragma unroll
    for (int t = 0; t < 12; ++t)
#pragma unroll
        for (int r = 0; r < 8; ++r) {
            const int row = q0 + half * 8 + r;
            obuf[((size_t)row * NH + head) * HD + t * 16 + l16] = acc[t][r] / l[r];
        }
}

// ---------------------------------------------------------------------------
// Host-side orchestration (graph-capture safe: only kernel launches).
// ---------------------------------------------------------------------------
extern "C" void kernel_launch(void* const* d_in, const int* in_sizes, int n_in,
                              void* d_out, int out_size, void* d_ws, size_t ws_size,
                              hipStream_t stream) {
    (void)in_sizes; (void)n_in; (void)out_size; (void)ws_size;

    const float* x         = (const float*)d_in[0];
    /* positions = arange(T): unused, index == token */
    const float* wq_a      = (const float*)d_in[2];   // [Q_LORA, HID]
    const float* q_norm_w  = (const float*)d_in[3];   // [Q_LORA]
    const float* wq_b      = (const float*)d_in[4];   // [H*D, Q_LORA]
    const float* wkv       = (const float*)d_in[5];   // [D, HID]
    const float* kv_norm_w = (const float*)d_in[6];   // [D]
    const float* attn_sink = (const float*)d_in[7];   // [H]
    const float* wo_a      = (const float*)d_in[8];   // [G*O_LORA, H*D/G]
    const float* wo_b      = (const float*)d_in[9];   // [HID, G*O_LORA]
    const float* rope_cos  = (const float*)d_in[10];  // [MAX_POS, DR/2]
    const float* rope_sin  = (const float*)d_in[11];
    float* y = (float*)d_out;                         // [T, HID] fp32

    // Workspace layout (fp32): ~132 MB total
    float* q_lora = (float*)d_ws;                               // [T, Q_LORA]
    float* qbuf   = q_lora + (size_t)T_TOK * Q_LORA;            // [T, H*D]
    float* kvbuf  = qbuf   + (size_t)T_TOK * NH * HD;           // [T, D]
    float* abuf   = kvbuf  + (size_t)T_TOK * HD;                // [T, H*D]
    float* olow   = abuf   + (size_t)T_TOK * NH * HD;           // [T, G*O_LORA]

    const dim3 gblk(32, 4);

    // 1) q_lora = x @ wq_a^T            [2048,1536], K=4096
    gemm_wmma_bf16<<<dim3(Q_LORA / 64, T_TOK / 16), gblk, 0, stream>>>(
        x, wq_a, q_lora, T_TOK, Q_LORA, HID, HID, HID, Q_LORA);
    // 2) RMSNorm(q_lora, q_norm_w)
    rmsnorm_kernel<<<T_TOK, 256, 0, stream>>>(q_lora, q_norm_w, Q_LORA);
    // 3) q = q_lora @ wq_b^T            [2048,6144], K=1536
    gemm_wmma_bf16<<<dim3((NH * HD) / 64, T_TOK / 16), gblk, 0, stream>>>(
        q_lora, wq_b, qbuf, T_TOK, NH * HD, Q_LORA, Q_LORA, Q_LORA, NH * HD);
    // 4) per-head unweighted RMSNorm over D (rows = T*H)
    rmsnorm_kernel<<<T_TOK * NH, 256, 0, stream>>>(qbuf, nullptr, HD);
    // 5) kv = x @ wkv^T                 [2048,192], K=4096
    gemm_wmma_bf16<<<dim3(HD / 64, T_TOK / 16), gblk, 0, stream>>>(
        x, wkv, kvbuf, T_TOK, HD, HID, HID, HID, HD);
    // 6) RMSNorm(kv, kv_norm_w)
    rmsnorm_kernel<<<T_TOK, 256, 0, stream>>>(kvbuf, kv_norm_w, HD);
    // 7) RoPE on q (rows = T*H, pos = row/H) and kv (pos = row)
    rope_kernel<<<T_TOK * NH, 32, 0, stream>>>(qbuf, rope_cos, rope_sin, NH, HD);
    rope_kernel<<<T_TOK, 32, 0, stream>>>(kvbuf, rope_cos, rope_sin, 1, HD);
    // 8) causal MQA attention with sink -> abuf [T,H,D]
    attn_wmma_kernel<<<dim3(T_TOK / 16, NH / 4), gblk, 0, stream>>>(
        qbuf, kvbuf, attn_sink, abuf);
    // 9) grouped low-rank: o_low[:, g*512:] = out[:, g*1536:] @ wo_a_g^T
    for (int g = 0; g < NG; ++g) {
        gemm_wmma_bf16<<<dim3(O_LORA / 64, T_TOK / 16), gblk, 0, stream>>>(
            abuf + (size_t)g * (NH * HD / NG),
            wo_a + (size_t)g * O_LORA * (NH * HD / NG),
            olow + (size_t)g * O_LORA,
            T_TOK, O_LORA, NH * HD / NG,
            NH * HD, NH * HD / NG, NG * O_LORA);
    }
    // 10) y = o_low @ wo_b^T            [2048,4096], K=2048
    gemm_wmma_bf16<<<dim3(HID / 64, T_TOK / 16), gblk, 0, stream>>>(
        olow, wo_b, y, T_TOK, HID, NG * O_LORA,
        NG * O_LORA, NG * O_LORA, HID);
}
